// Attention_6313601925349
// MI455X (gfx1250) — matrix operand order repaired and checked by reading, not compiled, not run on hardware
//
#include <hip/hip_runtime.h>
#include <hip/hip_bf16.h>

#define BSZ 2
#define TT 2048
#define DM 2048
#define NH 32
#define NKV 8
#define HD 64
#define NQKV 3072          // 2048 q + 512 k + 512 v rows of fused weight
#define ROWS (BSZ*TT)      // 4096

typedef __bf16 bf16;
typedef bf16 bf16x8 __attribute__((ext_vector_type(8)));
typedef bf16 v16bf __attribute__((ext_vector_type(16)));
typedef float v8f  __attribute__((ext_vector_type(8)));
typedef unsigned int u32x4 __attribute__((ext_vector_type(4)));
typedef int i32x8 __attribute__((ext_vector_type(8)));
typedef int i32x4 __attribute__((ext_vector_type(4)));

static __device__ __forceinline__ v16bf join16(bf16x8 lo, bf16x8 hi) {
  return __builtin_shufflevector(lo, hi, 0,1,2,3,4,5,6,7,8,9,10,11,12,13,14,15);
}

// ---------------- f32 -> bf16 convert ----------------
__global__ void cvt_bf16_kernel(const float* __restrict__ s, bf16* __restrict__ d, int n) {
  int i = blockIdx.x * blockDim.x + threadIdx.x;
  int stride = gridDim.x * blockDim.x;
  for (; i < n; i += stride) d[i] = (bf16)s[i];
}

// ---------------- bf16 WMMA GEMM: C[M,N] = A[M,K] * W[N,K]^T ----------------
// one wave computes a 32(M) x 64(N) tile; blockDim = 128 (4 waves).
// All addressing is uniform-base + 32-bit offset (GVS form) to keep address
// VGPR pressure low; unroll-by-2 lets the scheduler overlap clauses.
__global__ void gemm_bf16_wmma(const bf16* __restrict__ A, const bf16* __restrict__ W,
                               float* __restrict__ C, int M, int N, int K) {
  const int lane = threadIdx.x & 31;
  const int wv   = threadIdx.x >> 5;
  const int half = lane >> 4;
  const int ln   = lane & 15;
  const int tile   = blockIdx.x * 4 + wv;
  const int tilesN = N >> 6;
  const int m0 = (tile / tilesN) << 5;
  const int n0 = (tile % tilesN) << 6;

  const unsigned oa0 = (unsigned)(m0 + ln) * K + half * 16;
  const unsigned oa1 = (unsigned)(m0 + 16 + ln) * K + half * 16;
  const unsigned ow0 = (unsigned)(n0 + ln) * K + half * 16;
  const unsigned ow1 = (unsigned)(n0 + 16 + ln) * K + half * 16;
  const unsigned ow2 = (unsigned)(n0 + 32 + ln) * K + half * 16;
  const unsigned ow3 = (unsigned)(n0 + 48 + ln) * K + half * 16;

  v8f acc[2][4] = {};
#pragma unroll 2
  for (int k = 0; k < K; k += 32) {
    v16bf a0 = join16(*(const bf16x8*)(A + oa0 + k),
                      *(const bf16x8*)(A + oa0 + k + 8));
    v16bf a1 = join16(*(const bf16x8*)(A + oa1 + k),
                      *(const bf16x8*)(A + oa1 + k + 8));
    v16bf b0 = *(const v16bf*)(W + ow0 + k);
    v16bf b1 = *(const v16bf*)(W + ow1 + k);
    v16bf b2 = *(const v16bf*)(W + ow2 + k);
    v16bf b3 = *(const v16bf*)(W + ow3 + k);
    __builtin_prefetch(A + oa0 + k + 256, 0, 1);   // global_prefetch_b8
    acc[0][0] = __builtin_amdgcn_wmma_f32_16x16x32_bf16(false, a0, false, b0, (short)0, acc[0][0], false, false);
    acc[0][1] = __builtin_amdgcn_wmma_f32_16x16x32_bf16(false, a0, false, b1, (short)0, acc[0][1], false, false);
    acc[0][2] = __builtin_amdgcn_wmma_f32_16x16x32_bf16(false, a0, false, b2, (short)0, acc[0][2], false, false);
    acc[0][3] = __builtin_amdgcn_wmma_f32_16x16x32_bf16(false, a0, false, b3, (short)0, acc[0][3], false, false);
    acc[1][0] = __builtin_amdgcn_wmma_f32_16x16x32_bf16(false, a1, false, b0, (short)0, acc[1][0], false, false);
    acc[1][1] = __builtin_amdgcn_wmma_f32_16x16x32_bf16(false, a1, false, b1, (short)0, acc[1][1], false, false);
    acc[1][2] = __builtin_amdgcn_wmma_f32_16x16x32_bf16(false, a1, false, b2, (short)0, acc[1][2], false, false);
    acc[1][3] = __builtin_amdgcn_wmma_f32_16x16x32_bf16(false, a1, false, b3, (short)0, acc[1][3], false, false);
  }

#pragma unroll
  for (int i = 0; i < 2; ++i)
#pragma unroll
    for (int nb = 0; nb < 4; ++nb)
#pragma unroll
      for (int r = 0; r < 8; ++r)
        C[(unsigned)(m0 + i * 16 + r + 8 * half) * N + n0 + nb * 16 + ln] = acc[i][nb][r];
}

// ---------------- RoPE + relayout: qkv f32 (ROWS x 3072) -> Qb/Kb/Vt bf16 ----------------
__global__ void rope_layout_kernel(const float* __restrict__ qkv,
                                   const float* __restrict__ cosT,
                                   const float* __restrict__ sinT,
                                   bf16* __restrict__ Qb, bf16* __restrict__ Kb,
                                   bf16* __restrict__ Vt) {
  const int row = blockIdx.x;
  const int b = row >> 11;
  const int t = row & (TT - 1);
  const float* rp = qkv + (size_t)row * NQKV;
  for (int c = threadIdx.x; c < NQKV; c += blockDim.x) {
    float v = rp[c];
    if (c < 2048) {                      // Q with RoPE
      int h = c >> 6, d = c & 63;
      float cs = cosT[t * HD + d], sn = sinT[t * HD + d];
      float pv = rp[(d < 32) ? (c + 32) : (c - 32)];
      float o = (d < 32) ? (v * cs - pv * sn) : (v * cs + pv * sn);
      Qb[(unsigned)((b * NH + h) * TT + t) * HD + d] = (bf16)o;
    } else if (c < 2560) {               // K with RoPE
      int cc = c - 2048;
      int kv = cc >> 6, d = cc & 63;
      float cs = cosT[t * HD + d], sn = sinT[t * HD + d];
      float pv = rp[(d < 32) ? (c + 32) : (c - 32)];
      float o = (d < 32) ? (v * cs - pv * sn) : (v * cs + pv * sn);
      Kb[(unsigned)((b * NKV + kv) * TT + t) * HD + d] = (bf16)o;
    } else {                             // V, transposed (d-major)
      int cc = c - 2560;
      int kv = cc >> 6, d = cc & 63;
      Vt[(unsigned)((b * NKV + kv) * HD + d) * TT + t] = (bf16)v;
    }
  }
}

// ---------------- TDM helper: load a 32x64 bf16 K-tile into LDS ----------------
// 6-arg (clang-23 / therock) form: (g0, g1, g2, g3, g4, cpol)
static __device__ __forceinline__ void tdm_load_ktile(unsigned long long gaddr,
                                                      unsigned ldsOff) {
  u32x4 g0;
  g0[0] = 1u;                                            // count=1, load descriptor
  g0[1] = ldsOff;                                        // lds_addr (bytes)
  g0[2] = (unsigned)(gaddr & 0xffffffffu);               // global_addr[31:0]
  g0[3] = (unsigned)((gaddr >> 32) & 0x1ffffffu) | 0x80000000u; // addr[56:32] | type=2
  i32x8 g1;
  g1[0] = (int)(1u << 16);                               // data_size = 2 bytes
  g1[1] = (int)((unsigned)HD << 16);                     // tensor_dim0 = 64
  g1[2] = (int)((unsigned)TT << 16);                     // tensor_dim1 = 2048
  g1[3] = (int)((unsigned)HD << 16);                     // tile_dim0 = 64
  g1[4] = 32;                                            // tile_dim1 = 32 rows
  g1[5] = HD;                                            // tensor_dim0_stride = 64
  g1[6] = 0; g1[7] = 0;
  i32x4 z4 = {};
  i32x8 z8 = {};
  __builtin_amdgcn_tensor_load_to_lds(g0, g1, z4, z4, z8, 0);
}

// ---------------- flash attention: one wave = 32 queries, 32-key tiles ----------------
// K tiles arrive via TDM (double-buffered in LDS); V via contiguous global B-operand loads.
__global__ void attn_wmma_kernel(const bf16* __restrict__ Qb, const bf16* __restrict__ Kb,
                                 const bf16* __restrict__ Vt, bf16* __restrict__ Ob) {
  __shared__ __align__(16) bf16 ldsK[4][2][32 * 64];   // 32 KB: per-wave double buffer
  __shared__ __align__(16) bf16 pbuf[4][16 * 32];      // 4 KB: P transpose staging
  const int lane = threadIdx.x & 31;
  const int wv   = threadIdx.x >> 5;
  const int half = lane >> 4;
  const int ln   = lane & 15;
  const int tile = blockIdx.x * 4 + wv;  // 0..4095
  const int qt = tile & 63;              // 64 q-tiles of 32 rows
  const int bh = tile >> 6;
  const int b = bh >> 5, h = bh & 31;
  const int kv = h >> 2;                 // GQA: 4 q-heads per kv-head
  const int qbase = qt << 5;

  const unsigned qoff = (unsigned)(b * NH + h) * TT * HD;      // Q base offset
  const unsigned voff = (unsigned)(b * NKV + kv) * HD * TT;    // Vt base offset
  const bf16* Kp = Kb + (size_t)(b * NKV + kv) * TT * HD;

  // resident Q A-fragments for both 16-row groups
  v16bf aq[2][2];
#pragma unroll
  for (int rg = 0; rg < 2; ++rg) {
    const unsigned qr = qoff + (unsigned)(qbase + rg * 16 + ln) * HD;
    aq[rg][0] = join16(*(const bf16x8*)(Qb + qr + half * 16),
                       *(const bf16x8*)(Qb + qr + 8 + half * 16));
    aq[rg][1] = join16(*(const bf16x8*)(Qb + qr + 32 + half * 16),
                       *(const bf16x8*)(Qb + qr + 40 + half * 16));
  }

  float mrow[2][8], lrow[2][8];
  v8f acc[2][4] = {};
#pragma unroll
  for (int rg = 0; rg < 2; ++rg)
#pragma unroll
    for (int r = 0; r < 8; ++r) { mrow[rg][r] = -1.0e30f; lrow[rg][r] = 0.0f; }

  const float scale = 0.125f;            // 1/sqrt(64)
  const unsigned long long gK = (unsigned long long)(uintptr_t)Kp;
  const unsigned ldsOff0 = (unsigned)(uintptr_t)(&ldsK[wv][0][0]);
  const unsigned ldsOff1 = (unsigned)(uintptr_t)(&ldsK[wv][1][0]);

  // TDM prologue: fetch first K tile
  tdm_load_ktile(gK, ldsOff0);

  for (int jb = 0; jb <= qbase; jb += 32) {
    __builtin_amdgcn_s_wait_tensorcnt(0);
    // prefetch next K tile into the other buffer while we compute on this one
    if (jb + 32 <= qbase)
      tdm_load_ktile(gK + (unsigned long long)(jb + 32) * (HD * 2),
                     ((jb >> 5) & 1) ? ldsOff0 : ldsOff1);
    const bf16* kt = &ldsK[wv][(jb >> 5) & 1][0];

    // V B-operand fragments (shared by both row groups)
    v16bf vb[4];
#pragma unroll
    for (int nb = 0; nb < 4; ++nb)
      vb[nb] = *(const v16bf*)(Vt + voff + (unsigned)(nb * 16 + ln) * TT + jb + half * 16);

    // K^T B-operand fragments from LDS tile (row-major 32x64)
    v16bf kb00 = *(const v16bf*)(kt + (unsigned)ln * HD + half * 16);
    v16bf kb01 = *(const v16bf*)(kt + (unsigned)ln * HD + 32 + half * 16);
    v16bf kb10 = *(const v16bf*)(kt + (unsigned)(16 + ln) * HD + half * 16);
    v16bf kb11 = *(const v16bf*)(kt + (unsigned)(16 + ln) * HD + 32 + half * 16);

#pragma unroll
    for (int rg = 0; rg < 2; ++rg) {
      v8f S0 = {}, S1 = {};
      S0 = __builtin_amdgcn_wmma_f32_16x16x32_bf16(false, aq[rg][0], false, kb00, (short)0, S0, false, false);
      S0 = __builtin_amdgcn_wmma_f32_16x16x32_bf16(false, aq[rg][1], false, kb01, (short)0, S0, false, false);
      S1 = __builtin_amdgcn_wmma_f32_16x16x32_bf16(false, aq[rg][0], false, kb10, (short)0, S1, false, false);
      S1 = __builtin_amdgcn_wmma_f32_16x16x32_bf16(false, aq[rg][1], false, kb11, (short)0, S1, false, false);

      // online softmax across this 32-key tile
#pragma unroll
      for (int r = 0; r < 8; ++r) {
        const int q = qbase + rg * 16 + r + 8 * half;
        float s0 = S0[r] * scale;
        float s1 = S1[r] * scale;
        s0 = (jb + ln      > q) ? -1.0e30f : s0;   // causal mask
        s1 = (jb + 16 + ln > q) ? -1.0e30f : s1;
        float tmax = fmaxf(s0, s1);
#pragma unroll
        for (int msk = 1; msk < 16; msk <<= 1)
          tmax = fmaxf(tmax, __shfl_xor(tmax, msk, 32));
        const float mnew = fmaxf(mrow[rg][r], tmax);
        const float alpha = __expf(mrow[rg][r] - mnew);
        mrow[rg][r] = mnew;
        const float p0 = __expf(s0 - mnew);
        const float p1 = __expf(s1 - mnew);
        float rs = p0 + p1;
#pragma unroll
        for (int msk = 1; msk < 16; msk <<= 1)
          rs += __shfl_xor(rs, msk, 32);
        lrow[rg][r] = lrow[rg][r] * alpha + rs;
#pragma unroll
        for (int nb = 0; nb < 4; ++nb) acc[rg][nb][r] *= alpha;
        const int prow = (r + 8 * half) * 32;
        pbuf[wv][prow + ln]      = (bf16)p0;
        pbuf[wv][prow + 16 + ln] = (bf16)p1;
      }
      // reload P as A-fragment (16x32, K = 32 keys)
      v16bf pa = join16(*(const bf16x8*)&pbuf[wv][ln * 32 + half * 16],
                        *(const bf16x8*)&pbuf[wv][ln * 32 + 8 + half * 16]);
#pragma unroll
      for (int nb = 0; nb < 4; ++nb)
        acc[rg][nb] = __builtin_amdgcn_wmma_f32_16x16x32_bf16(
            false, pa, false, vb[nb], (short)0, acc[rg][nb], false, false);
    }
  }

  // normalize and write head-concat layout for the output projection
#pragma unroll
  for (int rg = 0; rg < 2; ++rg)
#pragma unroll
    for (int nb = 0; nb < 4; ++nb)
#pragma unroll
      for (int r = 0; r < 8; ++r) {
        const int q = qbase + rg * 16 + r + 8 * half;
        const float o = acc[rg][nb][r] / lrow[rg][r];
        Ob[(unsigned)(b * TT + q) * DM + h * HD + nb * 16 + ln] = (bf16)o;
      }
}

extern "C" void kernel_launch(void* const* d_in, const int* in_sizes, int n_in,
                              void* d_out, int out_size, void* d_ws, size_t ws_size,
                              hipStream_t stream) {
  (void)in_sizes; (void)n_in; (void)out_size;
  const float* x    = (const float*)d_in[0];
  const float* cosT = (const float*)d_in[1];
  const float* sinT = (const float*)d_in[2];
  const float* Wq   = (const float*)d_in[3];
  const float* Wk   = (const float*)d_in[4];
  const float* Wv   = (const float*)d_in[5];
  const float* Wo   = (const float*)d_in[6];
  float* out = (float*)d_out;

  char* ws = (char*)d_ws;
  const size_t o_xb   = 0;
  const size_t o_wqkv = o_xb   + (size_t)ROWS * DM * 2;
  const size_t o_wo   = o_wqkv + (size_t)NQKV * DM * 2;
  const size_t o_qkv  = o_wo   + (size_t)DM * DM * 2;
  const size_t o_q    = o_qkv  + (size_t)ROWS * NQKV * 4;
  const size_t o_k    = o_q    + (size_t)BSZ * NH * TT * HD * 2;
  const size_t o_v    = o_k    + (size_t)BSZ * NKV * TT * HD * 2;
  const size_t o_ao   = o_v    + (size_t)BSZ * NKV * TT * HD * 2;
  const size_t total  = o_ao   + (size_t)ROWS * DM * 2;
  if (ws_size < total) return;

  bf16* xb   = (bf16*)(ws + o_xb);
  bf16* wqkv = (bf16*)(ws + o_wqkv);
  bf16* wob  = (bf16*)(ws + o_wo);
  float* qkvf = (float*)(ws + o_qkv);
  bf16* Qb = (bf16*)(ws + o_q);
  bf16* Kb = (bf16*)(ws + o_k);
  bf16* Vt = (bf16*)(ws + o_v);
  bf16* aob = (bf16*)(ws + o_ao);

  cvt_bf16_kernel<<<4096, 256, 0, stream>>>(x,  xb,   ROWS * DM);
  cvt_bf16_kernel<<<2048, 256, 0, stream>>>(Wq, wqkv,             DM * DM);
  cvt_bf16_kernel<<<512,  256, 0, stream>>>(Wk, wqkv + (size_t)2048 * DM, 512 * DM);
  cvt_bf16_kernel<<<512,  256, 0, stream>>>(Wv, wqkv + (size_t)2560 * DM, 512 * DM);
  cvt_bf16_kernel<<<2048, 256, 0, stream>>>(Wo, wob,  DM * DM);

  // fused QKV projection: (4096 x 2048) * (3072 x 2048)^T
  gemm_bf16_wmma<<<(ROWS / 32) * (NQKV / 64) / 4, 128, 0, stream>>>(
      xb, wqkv, qkvf, ROWS, NQKV, DM);

  rope_layout_kernel<<<ROWS, 256, 0, stream>>>(qkvf, cosT, sinT, Qb, Kb, Vt);

  // causal GQA flash attention: 64 bh * 64 q-tiles = 4096 wave-tiles
  attn_wmma_kernel<<<(BSZ * NH * (TT / 32)) / 4, 128, 0, stream>>>(Qb, Kb, Vt, aob);

  // output projection: (4096 x 2048) * (2048 x 2048)^T -> f32 out
  gemm_bf16_wmma<<<(ROWS / 32) * (DM / 64) / 4, 128, 0, stream>>>(
      aob, wob, out, ROWS, DM, DM);
}
